// CouplingAttention_27084063768845
// MI455X (gfx1250) — matrix operand order repaired and checked by reading, not compiled, not run on hardware
//
#include <hip/hip_runtime.h>
#include <hip/hip_bf16.h>

// ---------------------------------------------------------------------------
// CouplingAttention on gfx1250. All contractions are D = A * B^T with
// A:[M,K], Bt:[N,K] row-major bf16 -> WMMA fragments are plain b128 loads.
// Register blocking: each wave computes a 32x64 output block (2x4 tiles of
// 16x16), so 6 fragment loads feed 8 v_wmma_f32_16x16x32_bf16 per K-step.
// ---------------------------------------------------------------------------

typedef __attribute__((ext_vector_type(8)))  __bf16 v8bf;
typedef __attribute__((ext_vector_type(16))) __bf16 v16bf;
typedef __attribute__((ext_vector_type(8)))  float  v8f;

#define DEVINL static __device__ __forceinline__
#define WMMA_BF16(a, b, c) \
  __builtin_amdgcn_wmma_f32_16x16x32_bf16(false, (a), false, (b), (short)0, (c), false, false)

DEVINL v16bf cat8(v8bf lo, v8bf hi) {
  return __builtin_shufflevector(lo, hi, 0,1,2,3,4,5,6,7,8,9,10,11,12,13,14,15);
}

// A fragment, 16x32 bf16 tile, row-major, leading dim ld (elements).
// Lanes 0-15: M=lane, elems 0..7 = K0..7, 8..15 = K16..23
// Lanes 16-31: M=lane-16, elems 0..7 = K8..15, 8..15 = K24..31
DEVINL v16bf frag_a(const __bf16* __restrict__ tile, int ld, int lane) {
  const __bf16* p = tile + (lane & 15) * ld + ((lane & 16) ? 8 : 0);
  return cat8(*(const v8bf*)p, *(const v8bf*)(p + 16));
}

// B fragment from B^T (N x K row-major): lane holds one column N with
// lanes 0-15: K0..7,K16..23; lanes 16-31: K8..15,K24..31.
DEVINL v16bf frag_bt(const __bf16* __restrict__ tile, int ld, int lane) {
  const __bf16* p = tile + (lane & 15) * ld + ((lane & 16) ? 8 : 0);
  return cat8(*(const v8bf*)p, *(const v8bf*)(p + 16));
}

// One K-step (K += 32) of the 32x64 register-blocked GEMM.
DEVINL void step_2x4(v8f (&acc)[2][4], const __bf16* __restrict__ Ak, int lda,
                     const __bf16* __restrict__ Btk, int ldb, int lane) {
  v16bf a0 = frag_a(Ak, lda, lane);
  v16bf a1 = frag_a(Ak + 16 * lda, lda, lane);
  v16bf b0 = frag_bt(Btk, ldb, lane);
  v16bf b1 = frag_bt(Btk + 16 * ldb, ldb, lane);
  v16bf b2 = frag_bt(Btk + 32 * ldb, ldb, lane);
  v16bf b3 = frag_bt(Btk + 48 * ldb, ldb, lane);
  acc[0][0] = WMMA_BF16(a0, b0, acc[0][0]);
  acc[0][1] = WMMA_BF16(a0, b1, acc[0][1]);
  acc[0][2] = WMMA_BF16(a0, b2, acc[0][2]);
  acc[0][3] = WMMA_BF16(a0, b3, acc[0][3]);
  acc[1][0] = WMMA_BF16(a1, b0, acc[1][0]);
  acc[1][1] = WMMA_BF16(a1, b1, acc[1][1]);
  acc[1][2] = WMMA_BF16(a1, b2, acc[1][2]);
  acc[1][3] = WMMA_BF16(a1, b3, acc[1][3]);
}

// Full K loop for simple row-major operands.
DEVINL void gemm_2x4(v8f (&acc)[2][4], const __bf16* __restrict__ A, int lda,
                     const __bf16* __restrict__ Bt, int ldb, int K, int lane) {
  for (int k0 = 0; k0 < K; k0 += 32) {
    if (k0 + 32 < K) {
      __builtin_prefetch(A + k0 + 32, 0, 1);
      __builtin_prefetch(Bt + k0 + 32, 0, 1);
    }
    step_2x4(acc, A + k0, lda, Bt + k0, ldb, lane);
  }
}

// ---------------------------------------------------------------------------
// Stage 0: conversions
// ---------------------------------------------------------------------------
__global__ __launch_bounds__(256) void cvt_bf16_kernel(const float* __restrict__ src,
                                                       __bf16* __restrict__ dst, int n) {
  int i = blockIdx.x * blockDim.x + threadIdx.x;
  int stride = gridDim.x * blockDim.x;
  for (; i < n; i += stride) dst[i] = (__bf16)src[i];
}

// W:[R,C] f32 -> Wt:[C,R] bf16
__global__ __launch_bounds__(256) void transpose_cvt_kernel(const float* __restrict__ W,
                                                            __bf16* __restrict__ Wt,
                                                            int R, int C) {
  int idx = blockIdx.x * blockDim.x + threadIdx.x;
  if (idx >= R * C) return;
  int oc = idx / R;
  int r  = idx - oc * R;
  Wt[idx] = (__bf16)W[r * C + oc];
}

// ---------------------------------------------------------------------------
// Stage 1: qkv = x @ Wqkv + b ; scatter q,k:[bh][t][w][d], v:[bh][t][d][w]
// M=65536, N=1536, K=512. Wave tile 32x64: 2048 x 24 wave-tiles.
// ---------------------------------------------------------------------------
__global__ __launch_bounds__(256) void qkv_gemm_kernel(const __bf16* __restrict__ xb,
                                                       const __bf16* __restrict__ WqkvT,
                                                       const float* __restrict__ bias,
                                                       __bf16* __restrict__ q,
                                                       __bf16* __restrict__ k,
                                                       __bf16* __restrict__ vt) {
  int lane = threadIdx.x & 31;
  int wav  = (blockIdx.x * blockDim.x + threadIdx.x) >> 5;
  int tn = wav % 24;          // 1536/64
  int tm = wav / 24;          // 65536/32
  const __bf16* A  = xb + tm * 32 * 512;
  const __bf16* Bt = WqkvT + tn * 64 * 512;
  v8f acc[2][4] = {};
  gemm_2x4(acc, A, 512, Bt, 512, 512, lane);

  int hi8 = (lane & 16) ? 8 : 0;
  for (int nt = 0; nt < 4; ++nt) {
    int o = tn * 64 + nt * 16 + (lane & 15);
    int s = o >> 9;           // 0:q 1:k 2:v
    int h = (o >> 6) & 7;
    int d = o & 63;
    float bo = bias[o];
    for (int mt = 0; mt < 2; ++mt) {
      int rbase = tm * 32 + mt * 16 + hi8;
      for (int r = 0; r < 8; ++r) {
        int m = rbase + r;
        int b = m >> 12;
        int n = m & 4095;
        int t = n >> 6, w = n & 63;
        int bh = b * 8 + h;
        __bf16 bv = (__bf16)(acc[mt][nt][r] + bo);
        if (s == 0)      q[((bh * 64 + t) * 64 + w) * 64 + d] = bv;
        else if (s == 1) k[((bh * 64 + t) * 64 + w) * 64 + d] = bv;
        else             vt[((bh * 64 + t) * 64 + d) * 64 + w] = bv;
      }
    }
  }
}

// ---------------------------------------------------------------------------
// Stage 2: row logits a[i,j] = SCALE * sum_{w,c} q[i,w,c] k[j,w,c]; K=4096.
// 2 waves per bh (one per 32-row half), full 64 columns per wave.
// ---------------------------------------------------------------------------
__global__ __launch_bounds__(256) void row_logits_kernel(const __bf16* __restrict__ q,
                                                         const __bf16* __restrict__ k,
                                                         float* __restrict__ logits) {
  const float kScale = 0.35355339059327373f;   // 64^-0.25
  int lane = threadIdx.x & 31;
  int wav  = (blockIdx.x * blockDim.x + threadIdx.x) >> 5;
  int bh   = wav >> 1;
  int half = wav & 1;
  const __bf16* A  = q + bh * 262144 + half * 32 * 4096;
  const __bf16* Bt = k + bh * 262144;
  v8f acc[2][4] = {};
  gemm_2x4(acc, A, 4096, Bt, 4096, 4096, lane);

  float* L = logits + bh * 4096;
  int hi8 = (lane & 16) ? 8 : 0;
  for (int mt = 0; mt < 2; ++mt)
    for (int nt = 0; nt < 4; ++nt) {
      int j = nt * 16 + (lane & 15);
      int ib = half * 32 + mt * 16 + hi8;
      for (int r = 0; r < 8; ++r) L[(ib + r) * 64 + j] = acc[mt][nt][r] * kScale;
    }
}

// ---------------------------------------------------------------------------
// Stage 3: col logits bm[i,j] = SCALE * sum_{t,c} q[t,i,c] k[t,j,c].
// Per 32-wide K block (t fixed, c contiguous) the common fragment loaders
// apply with ld=64 and base offset t*4096 + c0.
// ---------------------------------------------------------------------------
__global__ __launch_bounds__(256) void col_logits_kernel(const __bf16* __restrict__ q,
                                                         const __bf16* __restrict__ k,
                                                         float* __restrict__ logits) {
  const float kScale = 0.35355339059327373f;
  int lane = threadIdx.x & 31;
  int wav  = (blockIdx.x * blockDim.x + threadIdx.x) >> 5;
  int bh   = wav >> 1;
  int half = wav & 1;
  const __bf16* qb = q + bh * 262144 + half * 32 * 64;   // +i0*64
  const __bf16* kb = k + bh * 262144;
  v8f acc[2][4] = {};
  for (int k0 = 0; k0 < 4096; k0 += 32) {
    int tt = k0 >> 6;
    int c0 = k0 & 63;
    int off = tt * 4096 + c0;
    step_2x4(acc, qb + off, 64, kb + off, 64, lane);
  }
  float* L = logits + bh * 4096;
  int hi8 = (lane & 16) ? 8 : 0;
  for (int mt = 0; mt < 2; ++mt)
    for (int nt = 0; nt < 4; ++nt) {
      int j = nt * 16 + (lane & 15);
      int ib = half * 32 + mt * 16 + hi8;
      for (int r = 0; r < 8; ++r) L[(ib + r) * 64 + j] = acc[mt][nt][r] * kScale;
    }
}

// ---------------------------------------------------------------------------
// Stage 4: 64-wide softmax, fp32 logits -> bf16 probs. One thread per row.
// ---------------------------------------------------------------------------
__global__ __launch_bounds__(64) void softmax64_kernel(const float* __restrict__ logits,
                                                       __bf16* __restrict__ probs) {
  int g = blockIdx.x;
  int row = threadIdx.x;
  const float* L = logits + g * 4096 + row * 64;
  float mx = -3.0e38f;
  for (int j = 0; j < 64; ++j) mx = fmaxf(mx, L[j]);
  float s = 0.0f;
  for (int j = 0; j < 64; ++j) s += __expf(L[j] - mx);
  float inv = 1.0f / s;
  __bf16* P = probs + g * 4096 + row * 64;
  for (int j = 0; j < 64; ++j) P[j] = (__bf16)(__expf(L[j] - mx) * inv);
}

// ---------------------------------------------------------------------------
// Stage 5: tmp[(t,d),u] = sum_w vT[(t,d),w] * bm[u,w] -> tmp2[d][u][t] bf16.
// A = vT[bh]:[4096,64], Bt = bm[bh]:[64,64], K=64. 128 wave-tiles per bh.
// ---------------------------------------------------------------------------
__global__ __launch_bounds__(256) void tmp_gemm_kernel(const __bf16* __restrict__ vt,
                                                       const __bf16* __restrict__ bm,
                                                       __bf16* __restrict__ tmp2) {
  int lane = threadIdx.x & 31;
  int wav  = (blockIdx.x * blockDim.x + threadIdx.x) >> 5;
  int bh   = wav >> 7;
  int tm   = wav & 127;       // 4096/32
  const __bf16* A  = vt + bh * 262144 + tm * 32 * 64;
  const __bf16* Bt = bm + bh * 4096;
  v8f acc[2][4] = {};
  gemm_2x4(acc, A, 64, Bt, 64, 64, lane);

  __bf16* O = tmp2 + bh * 262144;
  int hi8 = (lane & 16) ? 8 : 0;
  for (int mt = 0; mt < 2; ++mt)
    for (int nt = 0; nt < 4; ++nt) {
      int u = nt * 16 + (lane & 15);
      int mb = tm * 32 + mt * 16 + hi8;
      for (int r = 0; r < 8; ++r) {
        int m = mb + r;
        int t = m >> 6, d = m & 63;
        O[d * 4096 + u * 64 + t] = (__bf16)acc[mt][nt][r];
      }
    }
}

// ---------------------------------------------------------------------------
// Stage 6: out[d][i,u] = sum_j a[i,j] * tmp2[d][u,j]; scatter into
// y[b*4096 + i*64+u][h*64+d] bf16. 2 wave-tiles per (bh,d).
// ---------------------------------------------------------------------------
__global__ __launch_bounds__(256) void out_gemm_kernel(const __bf16* __restrict__ a,
                                                       const __bf16* __restrict__ tmp2,
                                                       __bf16* __restrict__ y) {
  int lane = threadIdx.x & 31;
  int wav  = (blockIdx.x * blockDim.x + threadIdx.x) >> 5;
  int bh   = wav >> 7;
  int rest = wav & 127;
  int dd   = rest >> 1;
  int half = rest & 1;
  const __bf16* A  = a + bh * 4096 + half * 32 * 64;
  const __bf16* Bt = tmp2 + bh * 262144 + dd * 4096;
  v8f acc[2][4] = {};
  gemm_2x4(acc, A, 64, Bt, 64, 64, lane);

  int b = bh >> 3, h = bh & 7;
  int cc = h * 64 + dd;
  int hi8 = (lane & 16) ? 8 : 0;
  for (int mt = 0; mt < 2; ++mt)
    for (int nt = 0; nt < 4; ++nt) {
      int u = nt * 16 + (lane & 15);
      int ib = half * 32 + mt * 16 + hi8;
      for (int r = 0; r < 8; ++r) {
        int i = ib + r;
        y[(b * 4096 + i * 64 + u) * 512 + cc] = (__bf16)acc[mt][nt][r];
      }
    }
}

// ---------------------------------------------------------------------------
// Stage 7: final = y @ Wproj + b_proj, fp32 out. M=65536, N=512, K=512.
// 2048 x 8 wave-tiles of 32x64.
// ---------------------------------------------------------------------------
__global__ __launch_bounds__(256) void proj_gemm_kernel(const __bf16* __restrict__ y,
                                                        const __bf16* __restrict__ WprojT,
                                                        const float* __restrict__ bias,
                                                        float* __restrict__ out) {
  int lane = threadIdx.x & 31;
  int wav  = (blockIdx.x * blockDim.x + threadIdx.x) >> 5;
  int tn = wav & 7;           // 512/64
  int tm = wav >> 3;          // 65536/32
  const __bf16* A  = y + tm * 32 * 512;
  const __bf16* Bt = WprojT + tn * 64 * 512;
  v8f acc[2][4] = {};
  gemm_2x4(acc, A, 512, Bt, 512, 512, lane);

  int hi8 = (lane & 16) ? 8 : 0;
  for (int nt = 0; nt < 4; ++nt) {
    int o = tn * 64 + nt * 16 + (lane & 15);
    float bo = bias[o];
    for (int mt = 0; mt < 2; ++mt) {
      int mb = tm * 32 + mt * 16 + hi8;
      for (int r = 0; r < 8; ++r) out[(mb + r) * 512 + o] = acc[mt][nt][r] + bo;
    }
  }
}

// ---------------------------------------------------------------------------
// Host launch
// ---------------------------------------------------------------------------
extern "C" void kernel_launch(void* const* d_in, const int* in_sizes, int n_in,
                              void* d_out, int out_size, void* d_ws, size_t ws_size,
                              hipStream_t stream) {
  const float* x     = (const float*)d_in[0];   // [16,4096,512]
  const float* Wqkv  = (const float*)d_in[1];   // [512,1536]
  const float* bqkv  = (const float*)d_in[2];   // [1536]
  const float* Wproj = (const float*)d_in[3];   // [512,512]
  const float* bproj = (const float*)d_in[4];   // [512]
  float* out = (float*)d_out;

  char* w = (char*)d_ws;
  constexpr size_t OFF_WQKVT  = 0;                      // 1536*512*2
  constexpr size_t OFF_WPROJT = 1572864;                // 512*512*2
  constexpr size_t OFF_XB     = 2097152;                // 65536*512*2
  constexpr size_t OFF_Q      = OFF_XB + 67108864;
  constexpr size_t OFF_K      = OFF_Q + 67108864;
  constexpr size_t OFF_VT     = OFF_K + 67108864;
  constexpr size_t OFF_LOG    = OFF_VT + 67108864;      // 128*4096*4
  constexpr size_t OFF_A      = OFF_LOG + 2097152;      // 128*4096*2
  constexpr size_t OFF_BM     = OFF_A + 1048576;
  constexpr size_t OFF_TMP2   = OFF_XB;                 // alias (xb dead)
  constexpr size_t OFF_Y      = OFF_Q;                  // alias (q dead)

  __bf16* WqkvT  = (__bf16*)(w + OFF_WQKVT);
  __bf16* WprojT = (__bf16*)(w + OFF_WPROJT);
  __bf16* xb     = (__bf16*)(w + OFF_XB);
  __bf16* qb     = (__bf16*)(w + OFF_Q);
  __bf16* kb     = (__bf16*)(w + OFF_K);
  __bf16* vtb    = (__bf16*)(w + OFF_VT);
  float*  logit  = (float*) (w + OFF_LOG);
  __bf16* a_bf   = (__bf16*)(w + OFF_A);
  __bf16* bm_bf  = (__bf16*)(w + OFF_BM);
  __bf16* tmp2   = (__bf16*)(w + OFF_TMP2);
  __bf16* yb     = (__bf16*)(w + OFF_Y);

  // stage 0
  cvt_bf16_kernel<<<8192, 256, 0, stream>>>(x, xb, 65536 * 512);
  transpose_cvt_kernel<<<3072, 256, 0, stream>>>(Wqkv, WqkvT, 512, 1536);
  transpose_cvt_kernel<<<1024, 256, 0, stream>>>(Wproj, WprojT, 512, 512);

  // stage 1: 2048*24 = 49152 waves, 8 per block
  qkv_gemm_kernel<<<6144, 256, 0, stream>>>(xb, WqkvT, bqkv, qb, kb, vtb);

  // stage 2/4: row attention (256 waves)
  row_logits_kernel<<<32, 256, 0, stream>>>(qb, kb, logit);
  softmax64_kernel<<<128, 64, 0, stream>>>(logit, a_bf);

  // stage 3/4: column attention (reuses logit buffer)
  col_logits_kernel<<<32, 256, 0, stream>>>(qb, kb, logit);
  softmax64_kernel<<<128, 64, 0, stream>>>(logit, bm_bf);

  // stage 5: 128*128 = 16384 waves
  tmp_gemm_kernel<<<2048, 256, 0, stream>>>(vtb, bm_bf, tmp2);

  // stage 6: 128*64*2 = 16384 waves
  out_gemm_kernel<<<2048, 256, 0, stream>>>(a_bf, tmp2, yb);

  // stage 7: 2048*8 = 16384 waves
  proj_gemm_kernel<<<2048, 256, 0, stream>>>(yb, WprojT, bproj, out);
}
